// ContextAwareAttention_47991964566216
// MI455X (gfx1250) — matrix operand order repaired and checked by reading, not compiled, not run on hardware
//
#include <hip/hip_runtime.h>

#define DIM   1024
#define SEQ   2048
#define BATCH 8

typedef __attribute__((ext_vector_type(16))) __bf16       v16bf;
typedef __attribute__((ext_vector_type(8)))  float        v8f;
typedef __attribute__((ext_vector_type(4)))  unsigned int v4u;
typedef __attribute__((ext_vector_type(4)))  float        v4f;
typedef __attribute__((ext_vector_type(8)))  int          v8i;
typedef __attribute__((ext_vector_type(4)))  int          v4i;

union Frag { v16bf v; v4u q[2]; };

__device__ __forceinline__ unsigned short f2bf(float f) {
  unsigned u = __float_as_uint(f);
  u += 0x7FFFu + ((u >> 16) & 1u);       // round-to-nearest-even
  return (unsigned short)(u >> 16);
}

__device__ __forceinline__ v8f wmma_bf16(const Frag& a, const Frag& b, v8f c) {
  return __builtin_amdgcn_wmma_f32_16x16x32_bf16(
      /*neg_a=*/false, a.v, /*neg_b=*/false, b.v,
      /*c_mod=*/(short)0, c, /*reuse_a=*/false, /*reuse_b=*/false);
}

// TDM: stage `bytes` (multiple of 4, contiguous) from global -> LDS. One call
// per wave; tracked by TENSORcnt. Descriptor per CDNA5 ISA.
__device__ __forceinline__ void tdm_copy_to_lds(const void* gsrc, void* lds,
                                                unsigned bytes) {
  unsigned long long ga = (unsigned long long)(uintptr_t)gsrc;
  unsigned elems = bytes >> 2;           // data_size = 4B units
  v4u g0;
  g0[0] = 1u;                                            // count=1 (valid D#)
  g0[1] = (unsigned)(uintptr_t)lds;                      // lds_addr (bytes)
  g0[2] = (unsigned)(ga & 0xFFFFFFFFu);                  // global_addr[31:0]
  g0[3] = (unsigned)((ga >> 32) & 0x01FFFFFFu) | 0x80000000u; // [56:32]|type=2
  v8i g1;
  g1[0] = 0x20000;                       // workgroup_mask=0, data_size=2 (4B)
  g1[1] = (int)(elems << 16);            // tensor_dim0[15:0] @ bits 63:48
  g1[2] = (int)((elems >> 16) | (1u << 16)); // tensor_dim0[31:16], tensor_dim1=1
  g1[3] = (int)(elems << 16);            // tile_dim0 @ bits 127:112
  g1[4] = 1;                             // tile_dim1=1, tile_dim2=0
  g1[5] = (int)elems;                    // tensor_dim0_stride[31:0]
  g1[6] = 0;
  g1[7] = 0;
  v4i z4 = {0, 0, 0, 0};
#if __has_include(<hip/amd_detail/amd_gfx1250_TDM.h>)
  v8i z8 = {0, 0, 0, 0, 0, 0, 0, 0};
  __builtin_amdgcn_tensor_load_to_lds(g0, g1, z4, z4, z8, 0);
#else
  __builtin_amdgcn_tensor_load_to_lds(g0, g1, z4, z4, 0);
#endif
}

// ---------------------------------------------------------------------------
// Kernel 0: fp32 -> bf16 weight conversion for Wq, Wk, Wv
// ---------------------------------------------------------------------------
__global__ __launch_bounds__(256) void cvt_weights(
    const float* __restrict__ wq, const float* __restrict__ wk,
    const float* __restrict__ wv,
    unsigned short* __restrict__ oq, unsigned short* __restrict__ ok,
    unsigned short* __restrict__ ov) {
  int i = blockIdx.x * 256 + threadIdx.x;
  if (i < DIM * DIM) {
    oq[i] = f2bf(wq[i]);
    ok[i] = f2bf(wk[i]);
    ov[i] = f2bf(wv[i]);
  }
}

// ---------------------------------------------------------------------------
// Kernel 1: projection  Y = (X [+ context]) @ W^T + bias   (bf16 out)
//   transposeOut==0 : Y[m*D + n]             ([B][S][D])
//   transposeOut==1 : Y[(b*D + n)*S + s]     ([B][D][S], for V)
// Block: 256 thr (8 waves). Grid: (B*S/16, D/256). Wave -> two 16x16 tiles
// sharing one A fragment (register blocking).
// ---------------------------------------------------------------------------
__global__ __launch_bounds__(256) void proj_kernel(
    const float* __restrict__ X, const float* __restrict__ ctx,
    const unsigned short* __restrict__ Wb, const float* __restrict__ bias,
    unsigned short* __restrict__ Y, int addContext, int transposeOut) {
  __shared__ unsigned short sX[16 * DIM];  // 32 KB bf16 X tile

  const int m0 = blockIdx.x * 16;
  const int n0 = blockIdx.y * 256;
  const int tid = threadIdx.x;

  // Stage X tile (+context) as bf16 into LDS (needs fp32->bf16 transform,
  // so this path stays VALU; raw-copy paths below use the TDM instead).
  for (int i = tid * 4; i < 16 * DIM; i += 256 * 4) {
    int r = i >> 10, c = i & (DIM - 1);
    v4f x = *(const v4f*)(X + (size_t)(m0 + r) * DIM + c);
    if (addContext) {
      v4f cx = *(const v4f*)(ctx + c);
      x = x + cx;
    }
    sX[r * DIM + c + 0] = f2bf(x[0]);
    sX[r * DIM + c + 1] = f2bf(x[1]);
    sX[r * DIM + c + 2] = f2bf(x[2]);
    sX[r * DIM + c + 3] = f2bf(x[3]);
  }
  __syncthreads();

  const int wave = tid >> 5, lane = tid & 31;
  const int hf = lane >> 4, ln = lane & 15;
  const int na = n0 + wave * 32 + ln;        // first output column
  const int nb = na + 16;                    // second output column

  v8f acc0 = {}, acc1 = {};
  const unsigned short* wpa = Wb + (size_t)na * DIM;
  const unsigned short* wpb = Wb + (size_t)nb * DIM;
  for (int c = 0; c < DIM; c += 32) {
    Frag a, b0, b1;
    a.q[0]  = *(const v4u*)&sX[ln * DIM + c + hf * 16];
    a.q[1]  = *(const v4u*)&sX[ln * DIM + c + hf * 16 + 8];
    b0.q[0] = *(const v4u*)(wpa + c + hf * 16);
    b0.q[1] = *(const v4u*)(wpa + c + hf * 16 + 8);
    b1.q[0] = *(const v4u*)(wpb + c + hf * 16);
    b1.q[1] = *(const v4u*)(wpb + c + hf * 16 + 8);
    __builtin_prefetch(wpa + c + 64, 0, 1);
    __builtin_prefetch(wpb + c + 64, 0, 1);
    acc0 = wmma_bf16(a, b0, acc0);
    acc1 = wmma_bf16(a, b1, acc1);
  }

  const float bva = bias[na];
  const float bvb = bias[nb];
  for (int r = 0; r < 8; r++) {
    int row = m0 + r + hf * 8;               // global token index (C/D layout)
    unsigned short ya = f2bf(acc0[r] + bva);
    unsigned short yb = f2bf(acc1[r] + bvb);
    if (!transposeOut) {
      Y[(size_t)row * DIM + na] = ya;
      Y[(size_t)row * DIM + nb] = yb;
    } else {
      int bi = row >> 11, s = row & (SEQ - 1);
      Y[((size_t)bi * DIM + na) * SEQ + s] = ya;
      Y[((size_t)bi * DIM + nb) * SEQ + s] = yb;
    }
  }
}

// ---------------------------------------------------------------------------
// Kernel 2: causal attention for one (batch, 16-row query tile).
//   Qb,Kb: [B][S][D] bf16; VbT: [B][D][S] bf16; Out: [B][S][D] fp32
// LDS: scores f32 [16][S] (128K) + P bf16 [16][S] (64K) + Q bf16 (32K) = 224K.
// Q row-block staged once via TDM (tensor_load_to_lds) instead of 8 waves
// re-reading it from global.
// ---------------------------------------------------------------------------
__global__ __launch_bounds__(256) void attn_kernel(
    const unsigned short* __restrict__ Qb, const unsigned short* __restrict__ Kb,
    const unsigned short* __restrict__ VbT, float* __restrict__ Out) {
  extern __shared__ char smem[];
  float* sS = (float*)smem;                                     // [16][SEQ]
  unsigned short* sP = (unsigned short*)(smem + 16 * SEQ * 4);  // [16][SEQ]
  unsigned short* sQ = (unsigned short*)(smem + 16 * SEQ * 6);  // [16][DIM]

  const int qt = blockIdx.x;           // query tile 0..127
  const int b  = blockIdx.y;           // batch
  const int m0 = qt * 16;
  const int tid = threadIdx.x, wave = tid >> 5, lane = tid & 31;
  const int hf = lane >> 4, ln = lane & 15;

  const unsigned short* Qbase = Qb + (size_t)b * SEQ * DIM;
  const unsigned short* Kbase = Kb + (size_t)b * SEQ * DIM;
  const unsigned short* Vt    = VbT + (size_t)b * DIM * SEQ;
  const float scale = 0.03125f;        // D^-0.5 = 1/32

  // ---- Stage Q row-block (16 rows x DIM bf16 = 32 KB, contiguous) via TDM --
  if (wave == 0) {
    tdm_copy_to_lds(Qbase + (size_t)m0 * DIM, sQ, 16 * DIM * 2);
    __builtin_amdgcn_s_wait_tensorcnt(0);
  }
  __syncthreads();

  // ---- Step 1: S = scale * Q Kt, two key tiles per wave share the A frag ---
  for (int pt = wave * 2; pt <= qt; pt += 16) {
    const int kt0 = pt;
    const int kt1 = (pt + 1 <= qt) ? pt + 1 : pt;     // clamp tail (discarded)
    const unsigned short* kp0 = Kbase + (size_t)(kt0 * 16 + ln) * DIM;
    const unsigned short* kp1 = Kbase + (size_t)(kt1 * 16 + ln) * DIM;
    v8f acc0 = {}, acc1 = {};
    for (int c = 0; c < DIM; c += 32) {
      Frag a, b0, b1;
      a.q[0]  = *(const v4u*)&sQ[ln * DIM + c + hf * 16];
      a.q[1]  = *(const v4u*)&sQ[ln * DIM + c + hf * 16 + 8];
      b0.q[0] = *(const v4u*)(kp0 + c + hf * 16);
      b0.q[1] = *(const v4u*)(kp0 + c + hf * 16 + 8);
      b1.q[0] = *(const v4u*)(kp1 + c + hf * 16);
      b1.q[1] = *(const v4u*)(kp1 + c + hf * 16 + 8);
      acc0 = wmma_bf16(a, b0, acc0);
      acc1 = wmma_bf16(a, b1, acc1);
    }
    for (int r = 0; r < 8; r++) {
      int mm = r + hf * 8;             // row within tile (C/D layout)
      float s0 = acc0[r] * scale;
      if (kt0 == qt && ln > mm) s0 = -__builtin_inff();
      sS[mm * SEQ + kt0 * 16 + ln] = s0;
      if (pt + 1 <= qt) {
        float s1 = acc1[r] * scale;
        if (kt1 == qt && ln > mm) s1 = -__builtin_inff();
        sS[mm * SEQ + kt1 * 16 + ln] = s1;
      }
    }
  }
  __syncthreads();

  // ---- Step 2: row softmax (16 lanes cooperate per row) ----
  const int L = (qt + 1) * 16;               // valid key length
  const int Lpad = ((L + 31) / 32) * 32;     // pad to WMMA K granularity
  {
    int row = tid >> 4, sub = tid & 15;
    float mx = -__builtin_inff();
    for (int c = sub; c < L; c += 16) mx = fmaxf(mx, sS[row * SEQ + c]);
    for (int off = 8; off >= 1; off >>= 1) mx = fmaxf(mx, __shfl_xor(mx, off, 32));
    float ssum = 0.0f;
    for (int c = sub; c < L; c += 16) {
      float e = __expf(sS[row * SEQ + c] - mx);
      sS[row * SEQ + c] = e;
      ssum += e;
    }
    for (int off = 8; off >= 1; off >>= 1) ssum += __shfl_xor(ssum, off, 32);
    float inv = 1.0f / ssum;
    for (int c = sub; c < Lpad; c += 16)
      sP[row * SEQ + c] = (c < L) ? f2bf(sS[row * SEQ + c] * inv)
                                  : (unsigned short)0;
  }
  __syncthreads();

  // ---- Step 3: O = P V, two dim-tiles per iteration share the A frag ----
  const int nck = Lpad / 32;
  for (int j = 0; j < 4; j++) {
    int nt0 = wave * 2 + j * 16;       // tiles {2w, 2w+1} + 16j  -> covers 0..63
    int n0_ = nt0 * 16 + ln;
    int n1_ = n0_ + 16;
    const unsigned short* vp0 = Vt + (size_t)n0_ * SEQ;
    const unsigned short* vp1 = Vt + (size_t)n1_ * SEQ;
    v8f acc0 = {}, acc1 = {};
    for (int ck = 0; ck < nck; ck++) {
      int kb = ck * 32;
      Frag a, b0, b1;
      a.q[0]  = *(const v4u*)&sP[ln * SEQ + kb + hf * 16];
      a.q[1]  = *(const v4u*)&sP[ln * SEQ + kb + hf * 16 + 8];
      b0.q[0] = *(const v4u*)(vp0 + kb + hf * 16);
      b0.q[1] = *(const v4u*)(vp0 + kb + hf * 16 + 8);
      b1.q[0] = *(const v4u*)(vp1 + kb + hf * 16);
      b1.q[1] = *(const v4u*)(vp1 + kb + hf * 16 + 8);
      __builtin_prefetch(vp0 + kb + 64, 0, 1);
      __builtin_prefetch(vp1 + kb + 64, 0, 1);
      acc0 = wmma_bf16(a, b0, acc0);
      acc1 = wmma_bf16(a, b1, acc1);
    }
    for (int r = 0; r < 8; r++) {
      int q = m0 + r + hf * 8;
      Out[((size_t)b * SEQ + q) * DIM + n0_] = acc0[r];
      Out[((size_t)b * SEQ + q) * DIM + n1_] = acc1[r];
    }
  }
}

// ---------------------------------------------------------------------------
// Host launcher
// ---------------------------------------------------------------------------
extern "C" void kernel_launch(void* const* d_in, const int* in_sizes, int n_in,
                              void* d_out, int out_size, void* d_ws, size_t ws_size,
                              hipStream_t stream) {
  (void)in_sizes; (void)n_in; (void)out_size; (void)ws_size;
  const float* query = (const float*)d_in[0];
  const float* key_  = (const float*)d_in[1];
  const float* value = (const float*)d_in[2];
  const float* ctx   = (const float*)d_in[3];
  const float* Wq    = (const float*)d_in[4];
  const float* bq    = (const float*)d_in[5];
  const float* Wk    = (const float*)d_in[6];
  const float* bk    = (const float*)d_in[7];
  const float* Wv    = (const float*)d_in[8];
  const float* bv    = (const float*)d_in[9];
  float* out = (float*)d_out;

  char* ws = (char*)d_ws;
  const size_t QKV_BYTES = (size_t)BATCH * SEQ * DIM * 2;   // 32 MiB each
  const size_t W_BYTES   = (size_t)DIM * DIM * 2;           // 2 MiB each
  unsigned short* Qb  = (unsigned short*)(ws);
  unsigned short* Kb  = (unsigned short*)(ws + QKV_BYTES);
  unsigned short* VbT = (unsigned short*)(ws + 2 * QKV_BYTES);
  unsigned short* Wqb = (unsigned short*)(ws + 3 * QKV_BYTES);
  unsigned short* Wkb = (unsigned short*)(ws + 3 * QKV_BYTES + W_BYTES);
  unsigned short* Wvb = (unsigned short*)(ws + 3 * QKV_BYTES + 2 * W_BYTES);

  cvt_weights<<<(DIM * DIM + 255) / 256, 256, 0, stream>>>(Wq, Wk, Wv, Wqb, Wkb, Wvb);

  dim3 pg(BATCH * SEQ / 16, DIM / 256);
  proj_kernel<<<pg, 256, 0, stream>>>(query, ctx, Wqb, bq, Qb, 1, 0);
  proj_kernel<<<pg, 256, 0, stream>>>(key_,  ctx, Wkb, bk, Kb, 1, 0);
  proj_kernel<<<pg, 256, 0, stream>>>(value, ctx, Wvb, bv, VbT, 0, 1);

  const size_t shmem = (size_t)16 * SEQ * 6 + (size_t)16 * DIM * 2;  // 224 KiB
  (void)hipFuncSetAttribute(reinterpret_cast<const void*>(&attn_kernel),
                            hipFuncAttributeMaxDynamicSharedMemorySize,
                            (int)shmem);
  dim3 ag(SEQ / 16, BATCH);
  attn_kernel<<<ag, 256, shmem, stream>>>(Qb, Kb, VbT, out);
}
